// NeuralCDE4DSTEM_65429531787609
// MI455X (gfx1250) — compile-verified, matrix operand order checked by reading, not run on hardware
//
#include <hip/hip_runtime.h>
#include <hip/hip_bf16.h>

typedef __attribute__((ext_vector_type(16))) _Float16 v16h;
typedef __attribute__((ext_vector_type(8)))  _Float16 v8h;
typedef __attribute__((ext_vector_type(8)))  float    v8f;

#define NB     8
#define HW     16384
#define LATENT 256
#define TE_D   64
#define TQ     128
#define PHYS   130
#define NT     144      // 16 sparse + 128 query times
#define NSTEP  143
#define HPAD   480      // 450 padded to 15*32
#define PHK    288      // 257 padded to 9*32
#define HID    512

__device__ __forceinline__ float silu_f(float x) { return x / (1.0f + __expf(-x)); }

__device__ __forceinline__ v8f wmma16(v16h a, v16h b, v8f c) {
  return __builtin_amdgcn_wmma_f32_16x16x32_f16(false, a, false, b, (short)0, c, false, false);
}

// ---- fragment loaders (no branches, vector loads only) --------------------------------
// A fragment (16x32 at col k0) from f16 row-major [16 x ld]; ld and k0 multiples of 8.
// Lane l: m=l&15, kbase=k0+8*(l>>4); halfs 0..7 = K kbase.., halfs 8..15 = K kbase+16..
__device__ __forceinline__ v16h lda_rm(const _Float16* A, int ld, int k0, int lane) {
  int m  = lane & 15;
  int kb = k0 + ((lane >> 4) << 3);
  const v8h* p = (const v8h*)(A + (long)m * ld + kb);
  v8h lo = p[0];
  v8h hi = p[2];
  v16h r;
#pragma unroll
  for (int j = 0; j < 8; ++j) { r[j] = lo[j]; r[j + 8] = hi[j]; }
  return r;
}

// B fragment from pre-swizzled buffer: frag block = 512 halfs, lane-major (lane*16 + j).
__device__ __forceinline__ v16h ldfrag(const _Float16* Wsw, long frag, int lane) {
  return ((const v16h*)Wsw)[frag * 32 + lane];
}

// one 16x16 tile: A (f16 row-major, LDS or global) x pre-swizzled W
__device__ __forceinline__ v8f gemm_tile_sw(const _Float16* A, int ld, int ktiles,
                                            const _Float16* Wsw, int ntiles, int nt, int lane) {
  v8f c = {};
  for (int kt = 0; kt < ktiles; ++kt) {
    v16h a = lda_rm(A, ld, kt * 32, lane);
    v16h b = ldfrag(Wsw, (long)kt * ntiles + nt, lane);
    c = wmma16(a, b, c);
  }
  return c;
}

// C layout: VGPR r, lanes 0-15 -> M=r ; lanes 16-31 -> M=r+8 ; N = n0 + (lane&15)
__device__ __forceinline__ void store_act(v8f c, const float* bias, int N,
                                          _Float16* out, int ld, int n0, int lane, bool act) {
  int n  = n0 + (lane & 15);
  int mb = (lane >> 4) * 8;
  if (n < N) {
    float bn = bias[n];
#pragma unroll
    for (int r = 0; r < 8; ++r) {
      float v = c[r] + bn;
      if (act) v = silu_f(v);
      out[(r + mb) * ld + n] = (_Float16)v;
    }
  }
}

__device__ __forceinline__ void store_f32c(v8f c, const float* bias, float* out,
                                           int ld, int n0, int lane) {
  int n  = n0 + (lane & 15);
  int mb = (lane >> 4) * 8;
  float bn = bias[n];
#pragma unroll
  for (int r = 0; r < 8; ++r) out[(r + mb) * ld + n] = c[r] + bn;
}

// ---------------- prep: swizzle W[Klim x N] f32 -> WMMA B-fragment f16 layout -----------
__global__ void swizzle_b(const float* __restrict__ W, int Klim, int N,
                          int Ntiles, int total, _Float16* __restrict__ out) {
  int idx = blockIdx.x * blockDim.x + threadIdx.x;
  if (idx >= total) return;
  int e = idx & 511, frag = idx >> 9;
  int lane = e >> 4, j = e & 15;
  int kt = frag / Ntiles, nt = frag - kt * Ntiles;
  int n  = nt * 16 + (lane & 15);
  int kb = kt * 32 + ((lane >> 4) << 3);
  int k  = kb + ((j < 8) ? j : (8 + j));   // j>=8 -> 16 + (j-8)
  out[idx] = (k < Klim && n < N) ? (_Float16)W[(long)k * N + n] : (_Float16)0.0f;
}

// ---------------- prep: encoder A rows (8 real -> 16 padded) f32 -> f16 -----------------
__global__ void pad_a_enc(const float* __restrict__ seq, _Float16* __restrict__ out) {
  int idx = blockIdx.x * blockDim.x + threadIdx.x;
  if (idx >= 16 * HW) return;
  int m = idx >> 14, k = idx & (HW - 1);
  out[idx] = (m < NB) ? (_Float16)seq[(long)m * 16 * HW + k] : (_Float16)0.0f;  // b-stride K*H*W
}

// ---------------- times: stable sort by rank + searchsorted(left) -----------------------
__global__ void times_kernel(const float* __restrict__ st, const float* __restrict__ qt,
                             float* __restrict__ all_times, int* __restrict__ qidx) {
  __shared__ float v[NT];
  __shared__ float srt[NT];
  int b = blockIdx.x, tid = threadIdx.x;
  if (tid < NT) v[tid] = (tid < 16) ? st[b * 16 + tid] : qt[tid - 16];
  __syncthreads();
  if (tid < NT) {
    float x = v[tid];
    int rank = 0;
    for (int j = 0; j < NT; ++j) {
      float y = v[j];
      rank += (y < x) || (y == x && j < tid);
    }
    srt[rank] = x;
  }
  __syncthreads();
  if (tid < NT) all_times[b * NT + tid] = srt[tid];
  if (tid < TQ) {
    float q = qt[tid];
    int c = 0;
    for (int j = 0; j < NT; ++j) c += (srt[j] < q);
    qidx[b * TQ + tid] = c;
  }
}

// ---------------- encoder: z0 = Aenc @ enc_w + enc_b (K split over 4 waves) -------------
__global__ void encode_kernel(const _Float16* __restrict__ Aenc, const _Float16* __restrict__ enc_sw,
                              const float* __restrict__ enc_b, float* __restrict__ z0) {
  __shared__ float red[4 * 256];
  int tid = threadIdx.x, lane = tid & 31, wid = tid >> 5;
  int nt = blockIdx.x;
  v8f c = {};
  for (int kt = wid * 128; kt < wid * 128 + 128; ++kt) {
    v16h a = lda_rm(Aenc, HW, kt * 32, lane);
    v16h b = ldfrag(enc_sw, (long)kt * 16 + nt, lane);
    c = wmma16(a, b, c);
  }
#pragma unroll
  for (int r = 0; r < 8; ++r) red[wid * 256 + lane * 8 + r] = c[r];
  __syncthreads();
  if (wid == 0) {
    int n = nt * 16 + (lane & 15), mb = (lane >> 4) * 8;
#pragma unroll
    for (int r = 0; r < 8; ++r) {
      float s = red[lane * 8 + r] + red[256 + lane * 8 + r] +
                red[512 + lane * 8 + r] + red[768 + lane * 8 + r];
      int m = r + mb;
      if (m < NB) z0[m * LATENT + n] = s + enc_b[n];
    }
  }
}

// ---------------- ODE scan: one persistent workgroup, 16 waves, M=16 tile row -----------
__global__ void ode_kernel(const float* __restrict__ z0, const float* __restrict__ all_times,
                           const float* __restrict__ te_w1, const float* __restrict__ te_b1,
                           const float* __restrict__ te_w2, const float* __restrict__ te_b2,
                           const _Float16* __restrict__ ph_sw, const float* __restrict__ ph_b,
                           const _Float16* __restrict__ w0s, const float* __restrict__ b0,
                           const _Float16* __restrict__ w1s, const float* __restrict__ b1,
                           const _Float16* __restrict__ w2s, const float* __restrict__ b2,
                           const _Float16* __restrict__ w3s, const float* __restrict__ b3,
                           const _Float16* __restrict__ w4s, const float* __restrict__ b4,
                           float* __restrict__ traj) {
  extern __shared__ char smem[];
  float* zf   = (float*)smem;            // 16x256
  float* accz = zf + 16 * 256;
  float* kout = accz + 16 * 256;
  float* tf1  = kout + 16 * 256;         // 16x64
  float* t0v  = tf1 + 16 * 64;
  float* tmv  = t0v + 16;
  float* t1v  = tmv + 16;
  float* dtv  = t1v + 16;
  _Float16* hbuf = (_Float16*)(dtv + 16);    // 16x480 (960B rows, 16B aligned)
  _Float16* phin = hbuf + 16 * HPAD;         // 16x288
  _Float16* a1b  = phin + 16 * PHK;          // 16x512
  _Float16* a2b  = a1b + 16 * HID;           // 16x512

  int tid = threadIdx.x, lane = tid & 31, wid = tid >> 5;

  for (int i = tid; i < 16 * 256; i += 512) {
    int m = i >> 8;
    zf[i] = (m < NB) ? z0[i] : 0.0f;
    kout[i] = 0.0f;
    accz[i] = 0.0f;
  }
  for (int i = tid; i < 16 * HPAD; i += 512) hbuf[i] = (_Float16)0.0f;
  for (int i = tid; i < 16 * PHK; i += 512)  phin[i] = (_Float16)0.0f;
  for (int i = tid; i < NB * 256; i += 512) {
    int m = i >> 8, k = i & 255;
    traj[((long)m * NT) * 256 + k] = z0[i];
  }
  __syncthreads();

  for (int step = 0; step < NSTEP; ++step) {
    if (tid < 16) {
      float ta = 0.0f, tb = 0.0f;
      if (tid < NB) { ta = all_times[tid * NT + step]; tb = all_times[tid * NT + step + 1]; }
      t0v[tid] = ta; t1v[tid] = tb; tmv[tid] = 0.5f * (ta + tb); dtv[tid] = tb - ta;
    }
    __syncthreads();

    for (int s = 0; s < 4; ++s) {
      const float cin = (s == 0) ? 0.0f : ((s == 3) ? 1.0f : 0.5f);
      const float wk  = (s == 0 || s == 3) ? (1.0f / 6.0f) : (1.0f / 3.0f);
      const float* tcur = (s == 0) ? t0v : ((s == 3) ? t1v : tmv);

      // zin = z + cin*dt*k_prev  -> f16 into hbuf[0..255] and phin[0..255]
      for (int i = tid; i < 16 * 256; i += 512) {
        int m = i >> 8, k = i & 255;
        float zi = zf[i] + cin * dtv[m] * kout[i];
        _Float16 h = (_Float16)zi;
        hbuf[m * HPAD + k] = h;
        phin[m * PHK + k] = h;
      }
      if (tid < 16) phin[tid * PHK + 256] = (_Float16)tcur[tid];
      for (int i = tid; i < 16 * TE_D; i += 512) {
        int m = i >> 6, j = i & 63;
        tf1[i] = silu_f(tcur[m] * te_w1[j] + te_b1[j]);
      }
      __syncthreads();
      for (int i = tid; i < 16 * TE_D; i += 512) {
        int m = i >> 6, jj = i & 63;
        float a = te_b2[jj];
        for (int j = 0; j < TE_D; ++j) a += tf1[m * TE_D + j] * te_w2[j * TE_D + jj];
        hbuf[m * HPAD + 386 + jj] = (_Float16)a;
      }
      __syncthreads();
      // phys = silu([z,t] @ ph_w + ph_b) -> hbuf[256..385]  (9 N-tiles)
      if (wid < 9) {
        v8f c = gemm_tile_sw(phin, PHK, 9, ph_sw, 9, wid, lane);
        int n = wid * 16 + (lane & 15), mb = (lane >> 4) * 8;
        if (n < PHYS) {
          float bn = ph_b[n];
#pragma unroll
          for (int r = 0; r < 8; ++r)
            hbuf[(r + mb) * HPAD + 256 + n] = (_Float16)silu_f(c[r] + bn);
        }
      }
      __syncthreads();
      for (int t = wid; t < 32; t += 16) {
        v8f c = gemm_tile_sw(hbuf, HPAD, 15, w0s, 32, t, lane);
        store_act(c, b0, HID, a1b, HID, t * 16, lane, true);
      }
      __syncthreads();
      for (int t = wid; t < 32; t += 16) {
        v8f c = gemm_tile_sw(a1b, HID, 16, w1s, 32, t, lane);
        store_act(c, b1, HID, a2b, HID, t * 16, lane, true);
      }
      __syncthreads();
      for (int t = wid; t < 32; t += 16) {
        v8f c = gemm_tile_sw(a2b, HID, 16, w2s, 32, t, lane);
        store_act(c, b2, HID, a1b, HID, t * 16, lane, true);
      }
      __syncthreads();
      for (int t = wid; t < 32; t += 16) {
        v8f c = gemm_tile_sw(a1b, HID, 16, w3s, 32, t, lane);
        store_act(c, b3, HID, a2b, HID, t * 16, lane, true);
      }
      __syncthreads();
      {
        v8f c = gemm_tile_sw(a2b, HID, 16, w4s, 16, wid, lane);
        store_f32c(c, b4, kout, LATENT, wid * 16, lane);
      }
      __syncthreads();
      for (int i = tid; i < 16 * 256; i += 512) {
        int m = i >> 8;
        float term = wk * dtv[m] * kout[i];
        if (s == 0)      accz[i] = zf[i] + term;
        else if (s < 3)  accz[i] += term;
        else             zf[i] = accz[i] + term;
      }
      __syncthreads();
    }
    for (int i = tid; i < NB * 256; i += 512) {
      int m = i >> 8, k = i & 255;
      traj[((long)m * NT + step + 1) * 256 + k] = zf[m * 256 + k];
    }
    __syncthreads();
  }
}

// ---------------- decoder: z_query @ dec_w + dec_b (HBM-store bound) --------------------
__global__ void decode_kernel(const float* __restrict__ traj, const int* __restrict__ qidx,
                              const _Float16* __restrict__ dec_sw, const float* __restrict__ dec_b,
                              float* __restrict__ out) {
  __shared__ __align__(16) _Float16 As[16 * 256];
  int tid = threadIdx.x, lane = tid & 31, wid = tid >> 5;
  int bm = blockIdx.x;           // 64 M-tiles
  int bn = blockIdx.y;           // 128 groups of 8 N-tiles
  for (int i = tid; i < 16 * 256; i += 256) {
    int r = i >> 8, k = i & 255;
    int gm = bm * 16 + r;
    int b = gm >> 7, tq = gm & 127;
    const float* zr = traj + ((long)b * NT + qidx[b * TQ + tq]) * 256;
    As[r * 256 + k] = (_Float16)zr[k];
  }
  __syncthreads();
  int nt = bn * 8 + wid;
  v8f c = {};
  for (int kt = 0; kt < 8; ++kt) {
    v16h a = lda_rm(As, 256, kt * 32, lane);
    v16h b = ldfrag(dec_sw, (long)kt * 1024 + nt, lane);
    c = wmma16(a, b, c);
  }
  int n = nt * 16 + (lane & 15), mb = (lane >> 4) * 8;
  float bn2 = dec_b[n];
#pragma unroll
  for (int r = 0; r < 8; ++r) {
    int gm = bm * 16 + r + mb;
    out[(long)gm * HW + n] = c[r] + bn2;
  }
}

// ---------------- physics head: silu([z_query, t] @ ph_w + ph_b) ------------------------
__global__ void phys_kernel(const float* __restrict__ traj, const int* __restrict__ qidx,
                            const float* __restrict__ qt, const float* __restrict__ ph_w,
                            const float* __restrict__ ph_b, float* __restrict__ out) {
  int idx = blockIdx.x * blockDim.x + threadIdx.x;
  const int total = NB * TQ * PHYS;
  if (idx >= total) return;
  int n = idx % PHYS;
  int bt = idx / PHYS;
  int tq = bt % TQ, b = bt / TQ;
  const float* zr = traj + ((long)b * NT + qidx[b * TQ + tq]) * 256;
  float a = ph_b[n];
  for (int k = 0; k < 256; ++k) a += zr[k] * ph_w[k * PHYS + n];
  a += qt[tq] * ph_w[256 * PHYS + n];
  out[idx] = silu_f(a);
}

extern "C" void kernel_launch(void* const* d_in, const int* in_sizes, int n_in,
                              void* d_out, int out_size, void* d_ws, size_t ws_size,
                              hipStream_t stream) {
  const float* seq   = (const float*)d_in[0];
  const float* st    = (const float*)d_in[1];
  const float* qt    = (const float*)d_in[2];
  const float* enc_w = (const float*)d_in[3];
  const float* enc_b = (const float*)d_in[4];
  const float* te_w1 = (const float*)d_in[5];
  const float* te_b1 = (const float*)d_in[6];
  const float* te_w2 = (const float*)d_in[7];
  const float* te_b2 = (const float*)d_in[8];
  const float* ph_w  = (const float*)d_in[9];
  const float* ph_b  = (const float*)d_in[10];
  const float* w0    = (const float*)d_in[11];
  const float* b0    = (const float*)d_in[12];
  const float* w1    = (const float*)d_in[13];
  const float* b1    = (const float*)d_in[14];
  const float* w2    = (const float*)d_in[15];
  const float* b2    = (const float*)d_in[16];
  const float* w3    = (const float*)d_in[17];
  const float* b3    = (const float*)d_in[18];
  const float* w4    = (const float*)d_in[19];
  const float* b4    = (const float*)d_in[20];
  const float* dec_w = (const float*)d_in[21];
  const float* dec_b = (const float*)d_in[22];

  char* p = (char*)d_ws;
  auto carve = [&](size_t bytes) { char* r = p; p += (bytes + 255) & ~(size_t)255; return r; };

  float*    z0        = (float*)carve((size_t)NB * LATENT * 4);
  float*    all_times = (float*)carve((size_t)NB * NT * 4);
  int*      qidx      = (int*)carve((size_t)NB * TQ * 4);
  float*    traj      = (float*)carve((size_t)NB * NT * 256 * 4);
  _Float16* Aenc      = (_Float16*)carve((size_t)16 * HW * 2);
  _Float16* enc_sw    = (_Float16*)carve((size_t)512 * 16 * 512 * 2);
  _Float16* ph_sw     = (_Float16*)carve((size_t)9 * 9 * 512 * 2);
  _Float16* w0_sw     = (_Float16*)carve((size_t)15 * 32 * 512 * 2);
  _Float16* w1_sw     = (_Float16*)carve((size_t)16 * 32 * 512 * 2);
  _Float16* w2_sw     = (_Float16*)carve((size_t)16 * 32 * 512 * 2);
  _Float16* w3_sw     = (_Float16*)carve((size_t)16 * 32 * 512 * 2);
  _Float16* w4_sw     = (_Float16*)carve((size_t)16 * 16 * 512 * 2);
  _Float16* dec_sw    = (_Float16*)carve((size_t)8 * 1024 * 512 * 2);

  float* recon = (float*)d_out;                       // 8*128*16384
  float* physp = recon + (size_t)NB * TQ * HW;        // 8*128*130

  const int THR = 256;
  auto blk = [](int total) { return (total + 255) / 256; };

  // prep: weight swizzles + encoder A padding
  swizzle_b<<<blk(512 * 16 * 512), THR, 0, stream>>>(enc_w, HW, LATENT, 16, 512 * 16 * 512, enc_sw);
  swizzle_b<<<blk(9 * 9 * 512),    THR, 0, stream>>>(ph_w, 257, PHYS, 9, 9 * 9 * 512, ph_sw);
  swizzle_b<<<blk(15 * 32 * 512),  THR, 0, stream>>>(w0, 450, HID, 32, 15 * 32 * 512, w0_sw);
  swizzle_b<<<blk(16 * 32 * 512),  THR, 0, stream>>>(w1, HID, HID, 32, 16 * 32 * 512, w1_sw);
  swizzle_b<<<blk(16 * 32 * 512),  THR, 0, stream>>>(w2, HID, HID, 32, 16 * 32 * 512, w2_sw);
  swizzle_b<<<blk(16 * 32 * 512),  THR, 0, stream>>>(w3, HID, HID, 32, 16 * 32 * 512, w3_sw);
  swizzle_b<<<blk(16 * 16 * 512),  THR, 0, stream>>>(w4, HID, LATENT, 16, 16 * 16 * 512, w4_sw);
  swizzle_b<<<blk(8 * 1024 * 512), THR, 0, stream>>>(dec_w, LATENT, HW, 1024, 8 * 1024 * 512, dec_sw);
  pad_a_enc<<<blk(16 * HW), THR, 0, stream>>>(seq, Aenc);

  times_kernel<<<NB, 160, 0, stream>>>(st, qt, all_times, qidx);
  encode_kernel<<<16, 128, 0, stream>>>(Aenc, enc_sw, enc_b, z0);

  size_t lds = (size_t)(16 * 256 * 3 + 16 * 64 + 64) * 4 +
               (size_t)(16 * HPAD + 16 * PHK + 2 * 16 * HID) * 2;
  ode_kernel<<<1, 512, lds, stream>>>(z0, all_times, te_w1, te_b1, te_w2, te_b2,
                                      ph_sw, ph_b, w0_sw, b0, w1_sw, b1, w2_sw, b2,
                                      w3_sw, b3, w4_sw, b4, traj);

  decode_kernel<<<dim3(64, 128), 256, 0, stream>>>(traj, qidx, dec_sw, dec_b, recon);
  phys_kernel<<<blk(NB * TQ * PHYS), THR, 0, stream>>>(traj, qidx, qt, ph_w, ph_b, physp);
}
